// EncoderLSTM_47760036331637
// MI455X (gfx1250) — compile-verified, matrix operand order checked by reading, not run on hardware
//
#include <hip/hip_runtime.h>
#include <hip/hip_bf16.h>

// ---------------------------------------------------------------------------
// EncoderLSTM for MI455X (gfx1250): bf16 WMMA 16x16x32 (f32 accumulate),
// async global->LDS staging of the recurrent A panel, fused LSTM cell.
// T=512 B=256 XD=256 AD=64 H=1024 O=64, I=320, K_total = I+H = 1344 (42*32).
// ---------------------------------------------------------------------------

#define T_DIM 512
#define B_DIM 256
#define XD    256
#define AD    64
#define I_DIM 320      // XD + AD
#define H_DIM 1024
#define G_DIM 4096     // 4*H
#define K_TOT 1344     // I + H
#define O_DIM 64
#define A_LD  1352     // padded LDS leading dim (elems): row stride 676 dwords
                       // 676 mod 64 = 36 -> 16 rows hit 16 distinct banks

typedef __attribute__((ext_vector_type(16))) __bf16  v16bf;
typedef __attribute__((ext_vector_type(8)))  float   v8f;
typedef __attribute__((ext_vector_type(4)))  unsigned int u32x4;

union FragU { v16bf v; u32x4 q[2]; };

// float -> bf16 bits, round-to-nearest-even
__device__ __forceinline__ unsigned short f2bf(float f) {
    unsigned int u = __float_as_uint(f);
    u = (u + 0x7FFFu + ((u >> 16) & 1u)) >> 16;
    return (unsigned short)u;
}

__device__ __forceinline__ float sigm(float x) {
    return 1.0f / (1.0f + __expf(-x));
}

// Load one 16x32 bf16 fragment (A layout; B mirrors with lane = column).
// Per CDNA5 ISA: lanes 0-15 hold K 0..7 (elems 0-7) and 16..23 (elems 8-15);
// lanes 16-31 hold K 8..15 and 24..31.  Two b128 loads per lane.
__device__ __forceinline__ v16bf load_frag(const unsigned short* base, int ld, int k) {
    int lane = threadIdx.x & 31;
    int half = lane >> 4;       // 0 or 1
    int r    = lane & 15;       // row (A) / column (B)
    const unsigned short* p = base + (size_t)r * ld + k + half * 8;
    FragU f;
    f.q[0] = *(const u32x4*)(p);
    f.q[1] = *(const u32x4*)(p + 16);
    return f.v;
}

// ------------------------- pack / init kernels ------------------------------

__global__ void pack_y_kernel(const float* __restrict__ x,
                              const float* __restrict__ a,
                              const float* __restrict__ mask,
                              unsigned short* __restrict__ y) {
    long long idx = (long long)blockIdx.x * blockDim.x + threadIdx.x;
    const long long N = (long long)T_DIM * B_DIM * I_DIM;
    if (idx >= N) return;
    int k  = (int)(idx % I_DIM);
    long long tb = idx / I_DIM;            // t*B + b
    float v;
    if (k < XD) {
        long long o = tb * XD + k;
        v = x[o] * mask[o];
    } else {
        v = a[tb * AD + (k - XD)];
    }
    y[idx] = f2bf(v);
}

__global__ void pack_w_kernel(const float* __restrict__ W_ih,
                              const float* __restrict__ W_hh,
                              unsigned short* __restrict__ wcat) {
    long long idx = (long long)blockIdx.x * blockDim.x + threadIdx.x;
    const long long N = (long long)G_DIM * K_TOT;
    if (idx >= N) return;
    int k = (int)(idx % K_TOT);
    int n = (int)(idx / K_TOT);
    float v = (k < I_DIM) ? W_ih[(long long)n * I_DIM + k]
                          : W_hh[(long long)n * H_DIM + (k - I_DIM)];
    wcat[idx] = f2bf(v);
}

__global__ void pack_headw_kernel(const float* __restrict__ W_lin,
                                  const float* __restrict__ W_lv,
                                  unsigned short* __restrict__ hw) {
    int idx = blockIdx.x * blockDim.x + threadIdx.x;
    const int N = 2 * O_DIM * H_DIM;       // [128][1024]
    if (idx >= N) return;
    int k = idx % H_DIM;
    int n = idx / H_DIM;
    float v = (n < O_DIM) ? W_lin[n * H_DIM + k] : W_lv[(n - O_DIM) * H_DIM + k];
    hw[idx] = f2bf(v);
}

__global__ void bias_init_kernel(const float* __restrict__ b_ih,
                                 const float* __restrict__ b_hh,
                                 float* __restrict__ bias,
                                 unsigned short* __restrict__ h0,
                                 float* __restrict__ c) {
    int idx = blockIdx.x * blockDim.x + threadIdx.x;
    if (idx < G_DIM) bias[idx] = b_ih[idx] + b_hh[idx];
    if (idx < B_DIM * H_DIM) { h0[idx] = 0; c[idx] = 0.0f; }
}

// ------------------------- LSTM step kernel ---------------------------------
// grid = (32, 16): blockIdx.x -> hidden-column tile j0 = x*32 (32 cols of H)
//                  blockIdx.y -> batch tile m0 = y*16
// 8 waves: wave w computes gate g = w&3, sub-tile jsub = w>>2 (16 cols).
// A panel [16 batch rows x 1344 K] of [y_t | h] is staged into LDS once per
// step via gfx1250 async copies (ASYNCcnt), then all waves stream A-fragments
// from LDS (ds_load_b128) while B tiles of Wcat stream from L2.
// gates = A @ Wcat^T + bias, then fused LSTM cell update through LDS.

__global__ void __launch_bounds__(256)
lstm_step_kernel(const unsigned short* __restrict__ y_t,    // [B][320] bf16
                 const unsigned short* __restrict__ wcat,   // [4096][1344] bf16
                 const float* __restrict__ bias,            // [4096]
                 const unsigned short* __restrict__ h_src,  // [B][1024] bf16
                 unsigned short* __restrict__ h_dst,        // [B][1024] bf16
                 float* __restrict__ c_buf) {               // [B][1024] f32
    __shared__ __align__(16) unsigned short aTile[16 * A_LD];   // 43.3 KB
    __shared__ float gtile[8][16][16];                          // 8 KB

    const int lane = threadIdx.x & 31;
    const int w    = threadIdx.x >> 5;
    const int g    = w & 3;
    const int jsub = w >> 2;
    const int j0   = blockIdx.x * 32;
    const int m0   = blockIdx.y * 16;
    const int n0   = g * H_DIM + j0 + jsub * 16;

    // ---- stage A panel (16 rows of [y_t | h]) into LDS with async copies ----
    {
        const int GROUPS = K_TOT / 8;                // 168 x 16B chunks per row
        for (int idx = threadIdx.x; idx < 16 * GROUPS; idx += 256) {
            int r  = idx / GROUPS;
            int gk = idx - r * GROUPS;
            int k  = gk * 8;
            const unsigned short* src = (k < I_DIM)
                ? (y_t   + (size_t)(m0 + r) * I_DIM + k)
                : (h_src + (size_t)(m0 + r) * H_DIM + (k - I_DIM));
            unsigned lds_addr = (unsigned)(uintptr_t)(&aTile[r * A_LD + k]);
            unsigned long long gaddr = (unsigned long long)(uintptr_t)src;
            asm volatile("global_load_async_to_lds_b128 %0, %1, off"
                         :: "v"(lds_addr), "v"(gaddr) : "memory");
        }
        asm volatile("s_wait_asynccnt 0x0" ::: "memory");
    }
    __syncthreads();

    // ---- GEMM: init accumulator with bias (all 8 acc regs share column) ----
    float bv = bias[n0 + (lane & 15)];
    v8f acc;
    #pragma unroll
    for (int i = 0; i < 8; ++i) acc[i] = bv;

    const unsigned short* bW = wcat + (size_t)n0 * K_TOT;

    #pragma unroll 2
    for (int k = 0; k < K_TOT; k += 32) {
        v16bf af = load_frag(aTile, A_LD, k);     // ds_load_b128 x2
        v16bf bf = load_frag(bW, K_TOT, k);       // global_load_b128 x2
        acc = __builtin_amdgcn_wmma_f32_16x16x32_bf16(
                  false, af, false, bf, (short)0, acc, false, false);
    }

    // stash gate tile in LDS: row M = v + 8*(lane>>4), col N = lane&15
    {
        int row = 8 * (lane >> 4);
        int cn  = lane & 15;
        #pragma unroll
        for (int v = 0; v < 8; ++v) gtile[w][row + v][cn] = acc[v];
    }
    __syncthreads();

    // fused LSTM cell: 16 rows x 32 cols = 512 elements, 2 per thread
    #pragma unroll
    for (int e = threadIdx.x; e < 512; e += 256) {
        int row = e >> 5;           // 0..15 (batch within tile)
        int col = e & 31;           // 0..31 (hidden within tile)
        int js  = col >> 4;
        int cn  = col & 15;
        float xi = gtile[(js << 2) | 0][row][cn];
        float xf = gtile[(js << 2) | 1][row][cn];
        float xg = gtile[(js << 2) | 2][row][cn];
        float xo = gtile[(js << 2) | 3][row][cn];
        float ig = sigm(xi), fg = sigm(xf), gg = tanhf(xg), og = sigm(xo);
        int bi = m0 + row;
        int j  = j0 + col;
        size_t o = (size_t)bi * H_DIM + j;
        float cn_new = fg * c_buf[o] + ig * gg;
        c_buf[o] = cn_new;
        h_dst[o] = f2bf(og * tanhf(cn_new));
    }
}

// ------------------------- head kernel --------------------------------------
// mu = exp(h @ W_lin^T + b_lin)/10 ; log_var = h @ W_lv^T + b_lv - 5
// grid = 16 (batch tiles), 8 waves: wave w -> columns n0 = w*16 of [128].

__global__ void __launch_bounds__(256)
head_kernel(const unsigned short* __restrict__ h,     // [B][1024] bf16
            const unsigned short* __restrict__ hw,    // [128][1024] bf16
            const float* __restrict__ b_lin,
            const float* __restrict__ b_lv,
            float* __restrict__ out) {                // mu[256*64] ++ lv[256*64]
    const int lane = threadIdx.x & 31;
    const int w    = threadIdx.x >> 5;
    const int m0   = blockIdx.x * 16;
    const int n0   = w * 16;

    v8f acc = {};
    const unsigned short* aH = h  + (size_t)m0 * H_DIM;
    const unsigned short* bW = hw + (size_t)n0 * H_DIM;
    #pragma unroll 4
    for (int k = 0; k < H_DIM; k += 32) {
        v16bf af = load_frag(aH, H_DIM, k);
        v16bf bf = load_frag(bW, H_DIM, k);
        acc = __builtin_amdgcn_wmma_f32_16x16x32_bf16(
                  false, af, false, bf, (short)0, acc, false, false);
    }

    int n = n0 + (lane & 15);
    float badd = (n < O_DIM) ? b_lin[n] : b_lv[n - O_DIM];
    #pragma unroll
    for (int v = 0; v < 8; ++v) {
        int m = m0 + v + 8 * (lane >> 4);
        float val = acc[v] + badd;
        if (n < O_DIM) {
            out[(size_t)m * O_DIM + n] = __expf(val) * 0.1f;
        } else {
            out[(size_t)B_DIM * O_DIM + (size_t)m * O_DIM + (n - O_DIM)] = val - 5.0f;
        }
    }
}

// ------------------------- launcher -----------------------------------------

static inline size_t align256(size_t x) { return (x + 255) & ~(size_t)255; }

extern "C" void kernel_launch(void* const* d_in, const int* in_sizes, int n_in,
                              void* d_out, int out_size, void* d_ws, size_t ws_size,
                              hipStream_t stream) {
    (void)in_sizes; (void)n_in; (void)out_size; (void)ws_size;

    const float* x     = (const float*)d_in[0];
    const float* a     = (const float*)d_in[1];
    const float* mask  = (const float*)d_in[2];
    const float* W_ih  = (const float*)d_in[3];
    const float* W_hh  = (const float*)d_in[4];
    const float* b_ih  = (const float*)d_in[5];
    const float* b_hh  = (const float*)d_in[6];
    const float* W_lin = (const float*)d_in[7];
    const float* b_lin = (const float*)d_in[8];
    const float* W_lv  = (const float*)d_in[9];
    const float* b_lv  = (const float*)d_in[10];
    float* out = (float*)d_out;

    // workspace layout
    char* ws = (char*)d_ws;
    size_t off = 0;
    unsigned short* y    = (unsigned short*)(ws + off); off = align256(off + (size_t)T_DIM*B_DIM*I_DIM*2);
    unsigned short* wcat = (unsigned short*)(ws + off); off = align256(off + (size_t)G_DIM*K_TOT*2);
    float*          bias = (float*)         (ws + off); off = align256(off + (size_t)G_DIM*4);
    unsigned short* hb0  = (unsigned short*)(ws + off); off = align256(off + (size_t)B_DIM*H_DIM*2);
    unsigned short* hb1  = (unsigned short*)(ws + off); off = align256(off + (size_t)B_DIM*H_DIM*2);
    float*          cbuf = (float*)         (ws + off); off = align256(off + (size_t)B_DIM*H_DIM*4);
    unsigned short* hw   = (unsigned short*)(ws + off); off = align256(off + (size_t)2*O_DIM*H_DIM*2);

    // pack inputs to bf16
    {
        long long n = (long long)T_DIM * B_DIM * I_DIM;
        pack_y_kernel<<<(unsigned)((n + 255) / 256), 256, 0, stream>>>(x, a, mask, y);
    }
    {
        long long n = (long long)G_DIM * K_TOT;
        pack_w_kernel<<<(unsigned)((n + 255) / 256), 256, 0, stream>>>(W_ih, W_hh, wcat);
    }
    {
        int n = 2 * O_DIM * H_DIM;
        pack_headw_kernel<<<(n + 255) / 256, 256, 0, stream>>>(W_lin, W_lv, hw);
    }
    {
        int n = B_DIM * H_DIM;   // also covers G_DIM range for bias
        bias_init_kernel<<<(n + 255) / 256, 256, 0, stream>>>(b_ih, b_hh, bias, hb0, cbuf);
    }

    // reversed-time LSTM: step s processes t = T-1-s; h ping-pongs hb0<->hb1
    dim3 sgrid(H_DIM / 32, B_DIM / 16);   // (32, 16)
    for (int s = 0; s < T_DIM; ++s) {
        int t = T_DIM - 1 - s;
        const unsigned short* y_t  = y + (size_t)t * B_DIM * I_DIM;
        const unsigned short* hsrc = (s & 1) ? hb1 : hb0;
        unsigned short*       hdst = (s & 1) ? hb0 : hb1;
        lstm_step_kernel<<<sgrid, 256, 0, stream>>>(y_t, wcat, bias, hsrc, hdst, cbuf);
    }
    // after s = 511 (odd), final h is in hb0
    head_kernel<<<B_DIM / 16, 256, 0, stream>>>(hb0, hw, b_lin, b_lv, out);
}
